// SelfAtten_29283087024749
// MI455X (gfx1250) — matrix operand order repaired and checked by reading, not compiled, not run on hardware
//
#include <hip/hip_runtime.h>
#include <hip/hip_bf16.h>
#include <math.h>

typedef __attribute__((ext_vector_type(16))) _Float16 v16h;
typedef __attribute__((ext_vector_type(2)))  _Float16 h2;
typedef __attribute__((ext_vector_type(8)))  float    v8f;

#define PIMG 16384   // 128*128 pixels per image
#define NBAT 4
#define CDIM 192

__device__ inline float gelu_f(float x) {
    return 0.5f * x * (1.0f + erff(x * 0.70710678118654752f));
}

// packed-pair layout for GEMM B operands: element (b, c, p) of a C-channel tensor
// lives at [b][c/2][p][c&1]  ->  32-bit (k,k+1) pairs are contiguous per pixel
__device__ inline size_t pkidx(int b, int C, int c, int p) {
    return (((size_t)(b * (C >> 1) + (c >> 1))) << 15) + ((size_t)p << 1) + (c & 1);
}

// ---------------------------------------------------------------- copy f32
__global__ void copy_f32(const float* __restrict__ src, float* __restrict__ dst, int n) {
    int i = blockIdx.x * 256 + threadIdx.x;
    if (i < n) dst[i] = src[i];
}

// ---------------------------------------------------------------- f32 -> f16
__global__ void cvt_f16(const float* __restrict__ src, _Float16* __restrict__ dst, int n) {
    int i = blockIdx.x * 256 + threadIdx.x;
    if (i < n) dst[i] = (_Float16)src[i];
}

// ---------------------------------------------------------------- LayerNorm over C, NCHW f32 -> packed f16
__global__ void ln_nchw(const float* __restrict__ X, const float* __restrict__ w,
                        const float* __restrict__ bp, _Float16* __restrict__ Yp) {
    int idx = blockIdx.x * 256 + threadIdx.x;          // pixel id: 0..65535
    int b = idx >> 14, n = idx & 16383;
    size_t base = (((size_t)b * CDIM) << 14) + n;
    float s = 0.f, s2 = 0.f;
    for (int c = 0; c < CDIM; ++c) {
        float v = X[base + ((size_t)c << 14)];
        s += v; s2 += v * v;
    }
    float mu  = s * (1.0f / CDIM);
    float var = s2 * (1.0f / CDIM) - mu * mu;
    float rstd = rsqrtf(var + 1e-5f);
    for (int c = 0; c < CDIM; ++c) {
        float v = X[base + ((size_t)c << 14)];
        Yp[pkidx(b, CDIM, c, n)] = (_Float16)((v - mu) * rstd * w[c] + bp[c]);
    }
}

// ---------------------------------------------------------------- WMMA pointwise-conv GEMM
// out[b, o, p] = sum_c W[o,c] * Yp[b,c,p]   (Yp in packed-pair layout)
// one wave = 4 x (16 out_c x 16 pixel) tiles sharing one B fragment per K-step;
// K loop fully unrolled (CIN compile-time) for load/WMMA overlap
template <int CIN, bool RES>
__global__ __launch_bounds__(256) void gemm_pw(
        const _Float16* __restrict__ W, const _Float16* __restrict__ Yp,
        int Cout, _Float16* __restrict__ outH, float* __restrict__ outF,
        const float* __restrict__ res) {
    int lane = threadIdx.x & 31;
    int wave = threadIdx.x >> 5;
    int job  = blockIdx.x * 8 + wave;
    int nG = Cout >> 6;                 // groups of 4 tiles = 64 out channels
    int g  = job % nG;  job /= nG;
    int nt = job & 1023; int bb = job >> 10;
    int o0 = g << 6, p0 = nt << 4;
    bool lo = lane < 16;
    int mA = lane & 15;                 // A row (out channel within tile)
    int nB = lane & 15;                 // B/C column (pixel within tile)
    const _Float16* W0 = W + (size_t)(o0 + mA) * CIN;
    v8f acc[4] = {};
#pragma unroll
    for (int k0 = 0; k0 < CIN; k0 += 32) {
        // B 32x16 f16 fragment: lanes0-15 K=2v,2v+1 ; lanes16-31 K=16+2v,+1 ; N=lane&15
        v16h bm = {};
#pragma unroll
        for (int v = 0; v < 8; ++v) {
            int kb = 2 * v + (lo ? 0 : 16);
            size_t yi = (((size_t)(bb * (CIN >> 1) + ((k0 + kb) >> 1))) << 15)
                      + ((size_t)(p0 + nB) << 1);
            h2 pr = *(const h2*)(Yp + yi);          // coalesced 32-bit load
            bm[2 * v]     = pr[0];
            bm[2 * v + 1] = pr[1];
        }
#pragma unroll
        for (int j = 0; j < 4; ++j) {
            // A 16x32 f16 fragment: one contiguous 32B run per lane (K=16*(lane/16)+s)
            const _Float16* Wr = W0 + (size_t)(j * 16) * CIN + k0;
            v16h a = {};
#pragma unroll
            for (int v = 0; v < 8; ++v) {
                int ka = 2 * v + (lo ? 0 : 16);
                a[2 * v]     = Wr[ka];
                a[2 * v + 1] = Wr[ka + 1];
            }
            acc[j] = __builtin_amdgcn_wmma_f32_16x16x32_f16(
                         false, a, false, bm, (short)0, acc[j], false, false);
        }
    }
    // C layout: VGPR r -> M = r (+8 for upper lanes), N = lane&15
    int mC = lo ? 0 : 8;
#pragma unroll
    for (int j = 0; j < 4; ++j) {
#pragma unroll
        for (int r = 0; r < 8; ++r) {
            size_t oi = (((size_t)(bb * Cout + o0 + j * 16 + mC + r)) << 14) + p0 + nB;
            if (RES) outF[oi] = acc[j][r] + res[oi];
            else     outH[oi] = (_Float16)acc[j][r];
        }
    }
}

// ---------------------------------------------------------------- depthwise 3x3 SAME (opt. GELU in/out, opt. packed out)
__global__ void dw3x3(const _Float16* __restrict__ in, const float* __restrict__ wdw,
                      _Float16* __restrict__ out, int C, int geluIn, int geluOut, int packOut) {
    int idx = blockIdx.x * 256 + threadIdx.x;
    int n = idx & 16383;
    int rest = idx >> 14;
    int c = rest % C, b = rest / C;
    if (b >= NBAT) return;
    int h = n >> 7, w = n & 127;
    size_t base = ((size_t)(b * C + c)) << 14;
    const float* kw = wdw + (size_t)c * 9;
    float acc = 0.f;
#pragma unroll
    for (int dh = -1; dh <= 1; ++dh) {
        int ih = h + dh; if (ih < 0 || ih > 127) continue;
#pragma unroll
        for (int dw_ = -1; dw_ <= 1; ++dw_) {
            int iw = w + dw_; if (iw < 0 || iw > 127) continue;
            float v = (float)in[base + (ih << 7) + iw];
            if (geluIn) v = gelu_f(v);
            acc += v * kw[(dh + 1) * 3 + (dw_ + 1)];
        }
    }
    if (geluOut) acc = gelu_f(acc);
    if (packOut) out[pkidx(b, C, c, n)] = (_Float16)acc;
    else         out[base + n]          = (_Float16)acc;
}

// ---------------------------------------------------------------- channel attention (MDTA), one WG per (b,head)
// qkv: plain f16 [4][576][16384]; out: packed f16 [4][192][16384]
__global__ __launch_bounds__(768) void attn_mdta(const _Float16* __restrict__ qkv,
                                                 const float* __restrict__ tempv,
                                                 _Float16* __restrict__ outp) {
    int b = blockIdx.x >> 3, hH = blockIdx.x & 7;
    int tid = threadIdx.x;
    __shared__ float qs[24][132];
    __shared__ float ks[24][132];
    __shared__ float gram[24][24];
    __shared__ float qq[24], kk[24];
    __shared__ float attnS[24][24];
    size_t qb = ((size_t)(b * 576 +       hH * 24)) << 14;
    size_t kb = ((size_t)(b * 576 + 192 + hH * 24)) << 14;
    size_t vb = ((size_t)(b * 576 + 384 + hH * 24)) << 14;
    int c = tid / 24, d = tid % 24;
    float acc = 0.f, accq = 0.f, acck = 0.f;
    for (int n0 = 0; n0 < PIMG; n0 += 128) {
        for (int e = tid; e < 3072; e += 768) {
            int hc = e >> 7, j = e & 127;
            qs[hc][j] = (float)qkv[qb + ((size_t)hc << 14) + n0 + j];
            ks[hc][j] = (float)qkv[kb + ((size_t)hc << 14) + n0 + j];
        }
        __syncthreads();
        if (tid < 576) {
            for (int j = 0; j < 128; ++j) acc += qs[c][j] * ks[d][j];
        } else if (tid < 600) {
            int cc = tid - 576;
            for (int j = 0; j < 128; ++j) { float v = qs[cc][j]; accq += v * v; }
        } else if (tid < 624) {
            int cc = tid - 600;
            for (int j = 0; j < 128; ++j) { float v = ks[cc][j]; acck += v * v; }
        }
        __syncthreads();
    }
    if (tid < 576)      gram[c][d]    = acc;
    else if (tid < 600) qq[tid - 576] = accq;
    else if (tid < 624) kk[tid - 600] = acck;
    __syncthreads();
    if (tid < 24) {                            // softmax row (over d)
        float tmp = tempv[hH];
        float qn = fmaxf(sqrtf(qq[tid]), 1e-12f);
        float row[24]; float mx = -1e30f;
        for (int dd = 0; dd < 24; ++dd) {
            float kn = fmaxf(sqrtf(kk[dd]), 1e-12f);
            float av = gram[tid][dd] / (qn * kn) * tmp;
            row[dd] = av; mx = fmaxf(mx, av);
        }
        float ssum = 0.f;
        for (int dd = 0; dd < 24; ++dd) { row[dd] = expf(row[dd] - mx); ssum += row[dd]; }
        float inv = 1.0f / ssum;
        for (int dd = 0; dd < 24; ++dd) attnS[tid][dd] = row[dd] * inv;
    }
    __syncthreads();
    // out[c,n] = sum_d attn[c,d] * v[d,n]; threads = 24 channels x 32 lanes
    int co = tid >> 5, lane = tid & 31;
    int ch = hH * 24 + co;
    for (int n0 = 0; n0 < PIMG; n0 += 128) {
        for (int e = tid; e < 3072; e += 768) {
            int hc = e >> 7, j = e & 127;
            qs[hc][j] = (float)qkv[vb + ((size_t)hc << 14) + n0 + j];   // stage v
        }
        __syncthreads();
        for (int nn = lane; nn < 128; nn += 32) {
            float a2 = 0.f;
            for (int dd = 0; dd < 24; ++dd) a2 += attnS[co][dd] * qs[dd][nn];
            outp[pkidx(b, CDIM, ch, n0 + nn)] = (_Float16)a2;
        }
        __syncthreads();
    }
}

// ---------------------------------------------------------------- launch
extern "C" void kernel_launch(void* const* d_in, const int* in_sizes, int n_in,
                              void* d_out, int out_size, void* d_ws, size_t ws_size,
                              hipStream_t stream) {
    const float* xin    = (const float*)d_in[0];
    const float* ln1w   = (const float*)d_in[1];
    const float* ln1b   = (const float*)d_in[2];
    const float* wqkv   = (const float*)d_in[3];
    const float* dwqkv  = (const float*)d_in[4];
    const float* wproj  = (const float*)d_in[5];
    const float* temp   = (const float*)d_in[6];
    const float* ln2w   = (const float*)d_in[7];
    const float* ln2b   = (const float*)d_in[8];
    const float* ffnw1  = (const float*)d_in[9];
    const float* ffndw  = (const float*)d_in[10];
    const float* wout   = (const float*)d_in[11];
    float* X = (float*)d_out;                        // running fp32 state [4,192,128,128]

    const size_t NELT  = (size_t)NBAT * CDIM * PIMG;    // 12,582,912
    const size_t YB    = NELT * 2;                      // f16 bytes, 192-ch buffer
    const size_t QB    = (size_t)NBAT * 576 * PIMG * 2; // 576-ch f16 buffer
    char* ws = (char*)d_ws;
    _Float16* Yh = (_Float16*)(ws);                  // LN/attn out (packed) [4,96,16384,2]
    _Float16* Qh = (_Float16*)(ws + YB);             // qkv/ffn1 out (plain) [4,576,16384]
    _Float16* Dh = (_Float16*)(ws + YB + QB);        // dw out (plain or packed)
    _Float16* Wh = (_Float16*)(ws + YB + 2 * QB);    // f16 weights
    _Float16* WQKV  = Wh;                            // 576*192
    _Float16* WPROJ = Wh + 110592;                   // 192*192
    _Float16* WFFN1 = Wh + 147456;                   // 384*192
    _Float16* WOUT  = Wh + 221184;                   // 192*384

    copy_f32<<<(int)((NELT + 255) / 256), 256, 0, stream>>>(xin, X, (int)NELT);

    for (int i = 0; i < 2; ++i) {
        cvt_f16<<<(110592 + 255) / 256, 256, 0, stream>>>(wqkv  + (size_t)i * 110592, WQKV, 110592);
        cvt_f16<<<(36864  + 255) / 256, 256, 0, stream>>>(wproj + (size_t)i * 36864,  WPROJ, 36864);
        cvt_f16<<<(73728  + 255) / 256, 256, 0, stream>>>(ffnw1 + (size_t)i * 73728,  WFFN1, 73728);
        cvt_f16<<<(73728  + 255) / 256, 256, 0, stream>>>(wout  + (size_t)i * 73728,  WOUT,  73728);

        // --- attention half ---
        ln_nchw<<<256, 256, 0, stream>>>(X, ln1w + i * CDIM, ln1b + i * CDIM, Yh);
        // qkv GEMM: Cout=576 -> 9 groups * 1024 ptiles * 4 imgs / 8 waves
        gemm_pw<192, false><<<4608, 256, 0, stream>>>(WQKV, Yh, 576, Qh, nullptr, nullptr);
        dw3x3<<<147456, 256, 0, stream>>>(Qh, dwqkv + (size_t)i * 3 * CDIM * 9, Dh, 576, 0, 0, 0);
        attn_mdta<<<32, 768, 0, stream>>>(Dh, temp + i * 8, Yh);
        // proj GEMM + residual (fp32 in-place epilogue on X)
        gemm_pw<192, true><<<1536, 256, 0, stream>>>(WPROJ, Yh, 192, nullptr, X, X);

        // --- FFN half ---
        ln_nchw<<<256, 256, 0, stream>>>(X, ln2w + i * CDIM, ln2b + i * CDIM, Yh);
        gemm_pw<192, false><<<3072, 256, 0, stream>>>(WFFN1, Yh, 384, Qh, nullptr, nullptr);
        dw3x3<<<98304, 256, 0, stream>>>(Qh, ffndw + (size_t)i * 2 * CDIM * 9, Dh, 384, 1, 1, 1);
        // w_out GEMM (K=384, packed B from dw) + residual
        gemm_pw<384, true><<<1536, 256, 0, stream>>>(WOUT, Dh, 192, nullptr, X, X);
    }
}
